// CrossAttention_32280974197211
// MI455X (gfx1250) — matrix operand order repaired and checked by reading, not compiled, not run on hardware
//
#include <hip/hip_runtime.h>
#include <hip/hip_bf16.h>
#include <stdint.h>

// ---------------------------------------------------------------------------
// CDNA5 (gfx1250) cross-attention, bf16 WMMA (16x16x32) + f32 accumulate,
// with async global->LDS DMA staging (ASYNCcnt) and double-buffered tiles.
// Shapes fixed by reference: B=4, Sq=Skv=2048, H=16, D=64,
// QUERY_DIM=INNER=1024, CONTEXT_DIM=768.  Workspace: ~87 MiB.
// ---------------------------------------------------------------------------

typedef __bf16 v8bf  __attribute__((ext_vector_type(8)));
typedef __bf16 v16bf __attribute__((ext_vector_type(16)));
typedef float  v8f   __attribute__((ext_vector_type(8)));

__device__ __forceinline__ __bf16 f2bf(float f) {
  unsigned u = __builtin_bit_cast(unsigned, f);
  unsigned r = u + 0x7FFFu + ((u >> 16) & 1u);   // round-to-nearest-even
  return __builtin_bit_cast(__bf16, (unsigned short)(r >> 16));
}

__device__ __forceinline__ v16bf combine16(v8bf lo, v8bf hi) {
  v16bf r;
#pragma unroll
  for (int i = 0; i < 8; i++) { r[i] = lo[i]; r[i + 8] = hi[i]; }
  return r;
}
__device__ __forceinline__ v8f zero8() {
  v8f z;
#pragma unroll
  for (int i = 0; i < 8; i++) z[i] = 0.0f;
  return z;
}
__device__ __forceinline__ v8f wmma_bf16(v16bf a, v16bf b, v8f c) {
  // (neg_a, A, neg_b, B, c_mod, C, reuse_a, reuse_b)
  return __builtin_amdgcn_wmma_f32_16x16x32_bf16(false, a, false, b, (short)0,
                                                 c, false, false);
}

// Async DMA: 16 bytes per lane, global -> LDS, tracked with ASYNCcnt.
__device__ __forceinline__ void async_copy_b128(const __bf16* gptr,
                                                __bf16* lptr) {
  unsigned lds = (unsigned)(uintptr_t)lptr;               // addr[31:0] == LDS offset
  unsigned long long ga = (unsigned long long)(uintptr_t)gptr;
  asm volatile("global_load_async_to_lds_b128 %0, %1, off"
               :
               : "v"(lds), "v"(ga)
               : "memory");
}
__device__ __forceinline__ void wait_async0() {
  asm volatile("s_wait_asynccnt 0x0" ::: "memory");
}

// ---------------------------------------------------------------------------
// Prep kernel 1: f32 -> bf16, 8 elements/thread (vectorized).
// ---------------------------------------------------------------------------
__global__ __launch_bounds__(256)
void cvt_f32_bf16(const float* __restrict__ src, __bf16* __restrict__ dst) {
  size_t base = ((size_t)blockIdx.x * 256 + threadIdx.x) * 8;
  float4 a = *(const float4*)(src + base);
  float4 b = *(const float4*)(src + base + 4);
  v8bf o;
  o[0] = f2bf(a.x); o[1] = f2bf(a.y); o[2] = f2bf(a.z); o[3] = f2bf(a.w);
  o[4] = f2bf(b.x); o[5] = f2bf(b.y); o[6] = f2bf(b.z); o[7] = f2bf(b.w);
  *(v8bf*)(dst + base) = o;
}

// ---------------------------------------------------------------------------
// Prep kernel 2: transpose-convert weight W[K,N] (f32) -> Wt[N,K] (bf16).
// Block (32,8); 32x32 tile via LDS.  grid = (N/32, K/32).
// ---------------------------------------------------------------------------
__global__ __launch_bounds__(256)
void transpose_f32_bf16(const float* __restrict__ W, __bf16* __restrict__ Wt,
                        int K, int N) {
  __shared__ float t[32][33];
  const int n0 = blockIdx.x * 32, k0 = blockIdx.y * 32;
  const int tx = threadIdx.x, ty = threadIdx.y;
#pragma unroll
  for (int i = 0; i < 4; i++)
    t[ty + i * 8][tx] = W[(size_t)(k0 + ty + i * 8) * N + (n0 + tx)];
  __syncthreads();
#pragma unroll
  for (int i = 0; i < 4; i++)
    Wt[(size_t)(n0 + ty + i * 8) * K + (k0 + tx)] = f2bf(t[tx][ty + i * 8]);
}

// ---------------------------------------------------------------------------
// bf16 WMMA GEMM: C[M,N] = A[M,K] * Wt[N,K]^T   (block tile 128x128x32).
// Both operands already bf16; tiles staged via async DMA, double-buffered.
// MODE 0: store bf16 at [b, h, s, d]      (Q / K projections; scale folded)
// MODE 1: store bf16 at [b, h, d, s]      (V projection, transposed for PV)
// MODE 2: store f32  at [m, n] + bias[n]  (output projection)
// ---------------------------------------------------------------------------
template <int MODE>
__global__ __launch_bounds__(256)
void gemm_bf16_wmma(const __bf16* __restrict__ A, const __bf16* __restrict__ Wt,
                    const float* __restrict__ bias, void* __restrict__ outp,
                    int M, int N, int K, int S, float scale) {
  __shared__ __attribute__((aligned(16))) __bf16 As[2][128][40];  // [m][k]+pad
  __shared__ __attribute__((aligned(16))) __bf16 Bs[2][128][40];  // [n][k]+pad

  const int tid  = threadIdx.x;
  const int lane = tid & 31;
  const int w    = tid >> 5;
  const int half = lane >> 4;
  const int l15  = lane & 15;
  const int wm   = (w & 3) * 32;   // wave row offset within 128 tile
  const int wn   = (w >> 2) * 64;  // wave col offset within 128 tile
  const int mBase = blockIdx.y * 128;
  const int nBase = blockIdx.x * 128;

  // Issue async copies for one 128x32 A tile + 128x32 B tile (16B/lane x4).
  auto issue_tile = [&](int buf, int k0) {
#pragma unroll
    for (int i = 0; i < 2; i++) {
      int idx = i * 256 + tid;          // 0..511 -> 128 rows x 4 chunks
      int r = idx >> 2, c = (idx & 3) * 8;
      async_copy_b128(A + (size_t)(mBase + r) * K + k0 + c, &As[buf][r][c]);
    }
#pragma unroll
    for (int i = 0; i < 2; i++) {
      int idx = i * 256 + tid;
      int r = idx >> 2, c = (idx & 3) * 8;
      async_copy_b128(Wt + (size_t)(nBase + r) * K + k0 + c, &Bs[buf][r][c]);
    }
  };

  v8f acc[2][4];
#pragma unroll
  for (int mf = 0; mf < 2; mf++)
#pragma unroll
    for (int nf = 0; nf < 4; nf++) acc[mf][nf] = zero8();

  const int ntiles = K >> 5;
  issue_tile(0, 0);

  for (int kt = 0; kt < ntiles; kt++) {
    const int buf = kt & 1;
    wait_async0();       // this wave's DMA into buf complete
    __syncthreads();     // -> all waves' DMA into buf complete; all reads of
                         //    buf^1 from previous iteration are also done
    if (kt + 1 < ntiles) issue_tile(buf ^ 1, (kt + 1) * 32);

    // A-fragment: lane row = l15; e<8 -> k=half*8+e ; e>=8 -> k=16+half*8+e-8
    v16bf af[2];
#pragma unroll
    for (int mf = 0; mf < 2; mf++) {
      const __bf16* p = &As[buf][wm + mf * 16 + l15][half * 8];
      af[mf] = combine16(*(const v8bf*)p, *(const v8bf*)(p + 16));
    }
    // B-fragment: lane col = l15; e<8 -> k=half*8+e ; e>=8 -> k=16+half*8+e-8
#pragma unroll
    for (int nf = 0; nf < 4; nf++) {
      const __bf16* p = &Bs[buf][wn + nf * 16 + l15][half * 8];
      v16bf bf = combine16(*(const v8bf*)p, *(const v8bf*)(p + 16));
#pragma unroll
      for (int mf = 0; mf < 2; mf++)
        acc[mf][nf] = wmma_bf16(af[mf], bf, acc[mf][nf]);
    }
  }

  // C layout: lane holds col = l15, rows v + 8*half
#pragma unroll
  for (int mf = 0; mf < 2; mf++) {
#pragma unroll
    for (int nf = 0; nf < 4; nf++) {
#pragma unroll
      for (int v = 0; v < 8; v++) {
        int gm = mBase + wm + mf * 16 + v + 8 * half;
        int gn = nBase + wn + nf * 16 + l15;
        float val = acc[mf][nf][v] * scale;
        if (MODE == 2) {
          ((float*)outp)[(size_t)gm * N + gn] = val + bias[gn];
        } else {
          int b = gm / S, s2 = gm - b * S;
          int h = gn >> 6, d = gn & 63;
          if (MODE == 0)
            ((__bf16*)outp)[(((size_t)(b * 16 + h)) * S + s2) * 64 + d] =
                f2bf(val);
          else
            ((__bf16*)outp)[(((size_t)(b * 16 + h)) * 64 + d) * S + s2] =
                f2bf(val);
        }
      }
    }
  }
}

// ---------------------------------------------------------------------------
// Flash attention: Q [bh, Sq, 64] (pre-scaled), K [bh, Skv, 64], Vt [bh, 64, Skv]
// Each wave owns 16 query rows; 8 waves / block -> 128 q rows per block.
// Online softmax over Skv in tiles of 64. Output bf16 at [b, s, h*64+d].
// ---------------------------------------------------------------------------
__global__ __launch_bounds__(256)
void attn_fmha(const __bf16* __restrict__ Q, const __bf16* __restrict__ Km,
               const __bf16* __restrict__ Vt, __bf16* __restrict__ O) {
  constexpr int Sq = 2048, Skv = 2048, D = 64;
  __shared__ __attribute__((aligned(16))) __bf16 plds[8][16][72];

  const int tid  = threadIdx.x;
  const int lane = tid & 31;
  const int w    = tid >> 5;
  const int half = lane >> 4;
  const int l15  = lane & 15;
  const int qb   = blockIdx.x & 15;   // Sq/128 = 16 blocks per (b,h)
  const int bh   = blockIdx.x >> 4;
  const int q0   = qb * 128 + w * 16;

  const __bf16* Qp = Q + ((size_t)bh * Sq + q0) * D;
  const __bf16* Kp = Km + (size_t)bh * Skv * D;
  const __bf16* Vp = Vt + (size_t)bh * D * Skv;

  // Q A-fragments (fixed for the whole kv loop)
  v16bf aq[2];
#pragma unroll
  for (int ks = 0; ks < 2; ks++) {
    const __bf16* p = Qp + (size_t)l15 * D + ks * 32 + half * 8;
    aq[ks] = combine16(*(const v8bf*)p, *(const v8bf*)(p + 16));
  }

  v8f o[4];
#pragma unroll
  for (int df = 0; df < 4; df++) o[df] = zero8();
  float m[8], lsum[8];
#pragma unroll
  for (int v = 0; v < 8; v++) { m[v] = -3.0e38f; lsum[v] = 0.0f; }

  for (int j = 0; j < Skv / 64; j++) {
    const int kvb = j * 64;

    // S = Q * K^T  (B-frag element (k=d, n=kv) = K[kv][d], contiguous in d)
    v8f s[4];
#pragma unroll
    for (int nf = 0; nf < 4; nf++) {
      v8f c = zero8();
#pragma unroll
      for (int ks = 0; ks < 2; ks++) {
        const __bf16* p =
            Kp + (size_t)(kvb + nf * 16 + l15) * D + ks * 32 + half * 8;
        v16bf bk = combine16(*(const v8bf*)p, *(const v8bf*)(p + 16));
        c = wmma_bf16(aq[ks], bk, c);
      }
      s[nf] = c;
    }

    // Online softmax; lane owns rows v + 8*half, cols l15 across 16 lanes.
    float alpha[8];
#pragma unroll
    for (int v = 0; v < 8; v++) {
      float tm = s[0][v];
#pragma unroll
      for (int nf = 1; nf < 4; nf++) tm = fmaxf(tm, s[nf][v]);
      tm = fmaxf(tm, __shfl_xor(tm, 1, 32));
      tm = fmaxf(tm, __shfl_xor(tm, 2, 32));
      tm = fmaxf(tm, __shfl_xor(tm, 4, 32));
      tm = fmaxf(tm, __shfl_xor(tm, 8, 32));
      float nm = fmaxf(m[v], tm);
      alpha[v] = __expf(m[v] - nm);
      m[v] = nm;
      float rs = 0.0f;
#pragma unroll
      for (int nf = 0; nf < 4; nf++) {
        float p = __expf(s[nf][v] - nm);
        s[nf][v] = p;
        rs += p;
      }
      rs += __shfl_xor(rs, 1, 32);
      rs += __shfl_xor(rs, 2, 32);
      rs += __shfl_xor(rs, 4, 32);
      rs += __shfl_xor(rs, 8, 32);
      lsum[v] = lsum[v] * alpha[v] + rs;
    }
#pragma unroll
    for (int df = 0; df < 4; df++)
#pragma unroll
      for (int v = 0; v < 8; v++) o[df][v] *= alpha[v];

    // C-layout P -> A-layout bf16 via wave-private LDS (DS ops are in-order
    // within a wave; compiler inserts s_wait_dscnt for the RAW).
#pragma unroll
    for (int nf = 0; nf < 4; nf++)
#pragma unroll
      for (int v = 0; v < 8; v++)
        plds[w][v + 8 * half][nf * 16 + l15] = f2bf(s[nf][v]);

    v16bf ap[2];
#pragma unroll
    for (int ks = 0; ks < 2; ks++) {
      const __bf16* p = &plds[w][l15][ks * 32 + half * 8];
      ap[ks] = combine16(*(const v8bf*)p, *(const v8bf*)(p + 16));
    }

    // O += P * V   (B-frag element (k=kv, n=d) = Vt[d][kv], contiguous in kv)
#pragma unroll
    for (int df = 0; df < 4; df++) {
#pragma unroll
      for (int ks = 0; ks < 2; ks++) {
        const __bf16* p =
            Vp + (size_t)(df * 16 + l15) * Skv + kvb + ks * 32 + half * 8;
        v16bf bv = combine16(*(const v8bf*)p, *(const v8bf*)(p + 16));
        o[df] = wmma_bf16(ap[ks], bv, o[df]);
      }
    }
  }

  // Normalize and store to [b, s, h*64 + d] bf16 (A-input of final GEMM).
  const int b = bh >> 4, h = bh & 15;
#pragma unroll
  for (int v = 0; v < 8; v++) {
    float r = 1.0f / lsum[v];
    int q = q0 + v + 8 * half;
#pragma unroll
    for (int df = 0; df < 4; df++) {
      O[((size_t)(b * Sq + q)) * 1024 + h * 64 + df * 16 + l15] =
          f2bf(o[df][v] * r);
    }
  }
}

// ---------------------------------------------------------------------------
extern "C" void kernel_launch(void* const* d_in, const int* in_sizes, int n_in,
                              void* d_out, int out_size, void* d_ws,
                              size_t ws_size, hipStream_t stream) {
  (void)in_sizes; (void)n_in; (void)out_size; (void)ws_size;

  const float* x    = (const float*)d_in[0];  // [4,2048,1024]
  const float* ctx  = (const float*)d_in[1];  // [4,2048,768]
  const float* w_q  = (const float*)d_in[2];  // [1024,1024]
  const float* w_k  = (const float*)d_in[3];  // [768,1024]
  const float* w_v  = (const float*)d_in[4];  // [768,1024]
  const float* w_o  = (const float*)d_in[5];  // [1024,1024]
  const float* b_o  = (const float*)d_in[6];  // [1024]
  float* out = (float*)d_out;                 // [4,2048,1024] f32

  // Workspace layout (bf16 elements), ~87 MiB total.
  const size_t nx  = (size_t)8192 * 1024;   // x / attention-out (aliased)
  const size_t nc  = (size_t)8192 * 768;    // context
  const size_t nqq = (size_t)1024 * 1024;   // w_q / w_o transposed
  const size_t nkv = (size_t)1024 * 768;    // w_k / w_v transposed
  __bf16* xbf  = (__bf16*)d_ws;             // [8192,1024]; later: obuf
  __bf16* cbf  = xbf + nx;                  // [8192,768]
  __bf16* wqT  = cbf + nc;                  // [1024(N),1024(K)]
  __bf16* wkT  = wqT + nqq;                 // [1024(N),768(K)]
  __bf16* wvT  = wkT + nkv;                 // [1024(N),768(K)]
  __bf16* woT  = wvT + nkv;                 // [1024(N),1024(K)]
  __bf16* qbuf = woT + nqq;                 // [b,h,s,d], pre-scaled by 1/8
  __bf16* kbuf = qbuf + nx;                 // [b,h,s,d]
  __bf16* vbuf = kbuf + nx;                 // [b,h,d,s]  (transposed)
  __bf16* obuf = xbf;                       // alias: x dead after Q projection

  // --- Prep: convert activations, transpose-convert weights to bf16 ---
  cvt_f32_bf16<<<nx / (256 * 8), 256, 0, stream>>>(x, xbf);
  cvt_f32_bf16<<<nc / (256 * 8), 256, 0, stream>>>(ctx, cbf);
  dim3 tb(32, 8);
  transpose_f32_bf16<<<dim3(32, 32), tb, 0, stream>>>(w_q, wqT, 1024, 1024);
  transpose_f32_bf16<<<dim3(32, 24), tb, 0, stream>>>(w_k, wkT, 768, 1024);
  transpose_f32_bf16<<<dim3(32, 24), tb, 0, stream>>>(w_v, wvT, 768, 1024);
  transpose_f32_bf16<<<dim3(32, 32), tb, 0, stream>>>(w_o, woT, 1024, 1024);

  dim3 grid(1024 / 128, 8192 / 128);  // (N tiles, M tiles) = (8, 64)

  // Q = (x @ w_q) * scale   ; scale = 64^-0.5 = 0.125 (exact in bf16)
  gemm_bf16_wmma<0><<<grid, 256, 0, stream>>>(
      xbf, wqT, nullptr, qbuf, 8192, 1024, 1024, 2048, 0.125f);
  // K = ctx @ w_k
  gemm_bf16_wmma<0><<<grid, 256, 0, stream>>>(
      cbf, wkT, nullptr, kbuf, 8192, 1024, 768, 2048, 1.0f);
  // V^T = transpose(ctx @ w_v) per head
  gemm_bf16_wmma<1><<<grid, 256, 0, stream>>>(
      cbf, wvT, nullptr, vbuf, 8192, 1024, 768, 2048, 1.0f);
  // Flash attention: 4*16*(2048/128) = 1024 blocks, 8 waves each
  attn_fmha<<<1024, 256, 0, stream>>>(qbuf, kbuf, vbuf, obuf);
  // out = O @ w_o + b_o  (f32 output)
  gemm_bf16_wmma<2><<<grid, 256, 0, stream>>>(
      obuf, woT, b_o, out, 8192, 1024, 1024, 2048, 1.0f);
}
